// InstaNas_66580583022668
// MI455X (gfx1250) — matrix operand order repaired and checked by reading, not compiled, not run on hardware
//
#include <hip/hip_runtime.h>
#include <hip/hip_bf16.h>

typedef __attribute__((ext_vector_type(16))) _Float16 v16h;
typedef __attribute__((ext_vector_type(8)))  _Float16 v8h;
typedef __attribute__((ext_vector_type(8)))  float    v8f;

#define BATCH 64

#if __has_builtin(__builtin_amdgcn_sched_barrier)
#define SCHED_FENCE() __builtin_amdgcn_sched_barrier(0)
#else
#define SCHED_FENCE() do {} while (0)
#endif

// ---------------------------------------------------------------------------
// conv1: x [64,3,32,32] NCHW f32 -> h0 [64,32,32,32] NHWC f16 (BN + ReLU)
// ---------------------------------------------------------------------------
__global__ void conv1_kernel(const float* __restrict__ x, const float* __restrict__ w,
                             const float* __restrict__ scale, const float* __restrict__ bias,
                             _Float16* __restrict__ out) {
  int id = blockIdx.x * blockDim.x + threadIdx.x;  // ((b*32+y)*32+x)*32+co
  if (id >= BATCH * 32 * 32 * 32) return;
  int co = id & 31; int t = id >> 5;
  int xx = t & 31; t >>= 5;
  int yy = t & 31; t >>= 5;
  int b  = t;
  float acc = 0.f;
  for (int ci = 0; ci < 3; ++ci)
    for (int ry = 0; ry < 3; ++ry) {
      int iy = yy + ry - 1; if (iy < 0 || iy >= 32) continue;
      for (int rx = 0; rx < 3; ++rx) {
        int ix = xx + rx - 1; if (ix < 0 || ix >= 32) continue;
        acc += x[((b * 3 + ci) * 32 + iy) * 32 + ix] * w[((co * 3 + ci) * 3 + ry) * 3 + rx];
      }
    }
  float v = acc * scale[co] + bias[co];
  out[id] = (_Float16)fmaxf(v, 0.f);
}

// ---------------------------------------------------------------------------
// Pack f32 OIHW weights [K,Co,Ci,3,3] into per-lane WMMA A-fragment layout:
// dst[(((k*MT+mt)*KC + c)*32 + lane)*16 + e]  (MT=Co/16, KC=9*Ci/32)
// A 16x32 f16 as packed here (contiguous K): lane<16 holds row M=lane, K=0..15;
//                     lane>=16 holds row M=lane-16, K=16..31.
// K-dim ordering: kd = tap*Ci + ci, tap = ry*3+rx.
// ---------------------------------------------------------------------------
__global__ void pack_w_kernel(const float* __restrict__ w, _Float16* __restrict__ dst,
                              int Co, int Ci) {
  int KC = 9 * Ci / 32;
  long total = 4L * Co * Ci * 9;
  long id = (long)blockIdx.x * blockDim.x + threadIdx.x;
  if (id >= total) return;
  int e = id & 15; long t = id >> 4;
  int lane = t & 31; t >>= 5;
  int c  = (int)(t % KC); t /= KC;
  int mt = (int)(t % (Co / 16)); t /= (Co / 16);
  int k  = (int)t;
  int base = (lane < 16) ? 0 : 16;
  int kk = base + e;
  int co = mt * 16 + (lane & 15);
  int kd = c * 32 + kk;
  int tap = kd / Ci, ci = kd % Ci;
  int ry = tap / 3, rx = tap % 3;
  dst[id] = (_Float16)w[((long)(k * Co + co) * Ci + ci) * 9 + ry * 3 + rx];
}

// B fragment (32x16 activations): lane<16 -> col=lane, K=ci0+0..15;
// lane>=16 -> col=lane-16, K=ci0+16..31. One 32B contiguous load, zero-padded.
__device__ __forceinline__ v16h load_bfrag(const _Float16* hp, int cc, bool valid) {
  v16h bf;
  for (int e = 0; e < 16; ++e) bf[e] = (_Float16)0.f;
  if (valid) bf = *(const v16h*)(hp + cc * 32);
  return bf;
}

// ---------------------------------------------------------------------------
// Fused block (templated geometry): per branch k, implicit-GEMM conv via
// v_wmma_f32_16x16x32_f16. One wave: 64 Co x 16 pixels (4 C-tiles).
// The (tap x chunk) loop is flattened and fully unrolled; fragments for chunk
// c+1 are prefetched (fresh SSA values) and a sched_barrier pins them above
// the WMMAs of chunk c, so waits become partial and the VMEM stream overlaps
// the matrix pipe across tap boundaries too.
// ---------------------------------------------------------------------------
template <int Ci, int Co, int Hin, int Win, int Hout, int Wout, int STRIDE, int IDX0>
__global__ __launch_bounds__(256) void block_gemm_kernel(
    const _Float16* __restrict__ hin, const _Float16* __restrict__ wpack,
    const float* __restrict__ bias, const int* __restrict__ policy,
    _Float16* __restrict__ hout, int t) {
  constexpr int MT64 = Co / 64;        // 64-channel groups (power of two)
  constexpr int HW   = Hout * Wout;    // power of two
  constexpr int Npix = BATCH * HW;
  constexpr int NT   = Npix / 16;
  constexpr int CPT  = Ci / 32;        // 32-wide K-chunks per tap
  constexpr int KC   = 9 * CPT;
  constexpr int MT   = Co / 16;

  const int lane = threadIdx.x & 31;
  const int wid  = blockIdx.x * 8 + (threadIdx.x >> 5);
  if (wid >= MT64 * NT) return;        // wave-uniform guard
  const int cg = wid & (MT64 - 1);
  const int pt = wid / MT64;

  // Per-lane output pixel (a 16-pixel tile never crosses a batch element)
  const int p   = pt * 16 + (lane & 15);
  const int b   = p / HW;
  const int rem = p & (HW - 1);
  const int y   = rem / Wout;
  const int x   = rem & (Wout - 1);
  const int laneoff = (lane < 16) ? 0 : 16;

  // Policy is wave-uniform (same b across the tile); force scalar so branch
  // skipping keeps EXEC all-ones around the WMMAs.
  int aw[4]; int any = 0;
  for (int k = 0; k < 4; ++k) {
    aw[k] = __builtin_amdgcn_readfirstlane(policy[(b * 6 + t) * 4 + k]);
    any |= aw[k];
  }
  const bool no_act = (any == 0);

  // Precompute per-tap activation row pointers + validity once per wave.
  // Constant-indexed after unroll -> stays in registers.
  const _Float16* hpt[9];
  bool vld[9];
#pragma unroll
  for (int tap = 0; tap < 9; ++tap) {
    const int ry = tap / 3, rx = tap % 3;
    const int iy = y * STRIDE + ry - 1;
    const int ix = x * STRIDE + rx - 1;
    vld[tap] = ((unsigned)iy < (unsigned)Hin) && ((unsigned)ix < (unsigned)Win);
    hpt[tap] = hin + (((size_t)b * Hin + iy) * Win + ix) * Ci + laneoff;
  }

  v8f out[4];
  for (int m = 0; m < 4; ++m)
    for (int r = 0; r < 8; ++r) out[m][r] = 0.f;

  for (int k = 0; k < 4; ++k) {
    float wk = (float)aw[k];
    if (IDX0 && no_act && k == 2) wk = 1.f;   // feats[2] fallback via weight trick
    if (wk == 0.f) continue;                  // uniform skip

    v8f acc[4];
    for (int m = 0; m < 4; ++m)
      for (int r = 0; r < 8; ++r) acc[m][r] = 0.f;

    const _Float16* wptr = wpack + ((size_t)(k * MT + cg * 4) * KC) * 512 + (size_t)lane * 16;

    // ---- pipeline prologue: chunk 0 ----
    v16h bf = load_bfrag(hpt[0], 0, vld[0]);
    v16h a0 = *(const v16h*)(wptr + (size_t)0 * KC * 512);
    v16h a1 = *(const v16h*)(wptr + (size_t)1 * KC * 512);
    v16h a2 = *(const v16h*)(wptr + (size_t)2 * KC * 512);
    v16h a3 = *(const v16h*)(wptr + (size_t)3 * KC * 512);

#pragma unroll
    for (int c = 0; c < KC; ++c) {
      v16h nbf, na0, na1, na2, na3;
      if (c + 1 < KC) {                          // folds at compile time
        const int tap = (c + 1) / CPT;           // constants after unroll
        const int cc  = (c + 1) - tap * CPT;
        nbf = load_bfrag(hpt[tap], cc, vld[tap]);
        const _Float16* wn = wptr + (size_t)(c + 1) * 512;
        na0 = *(const v16h*)(wn + (size_t)0 * KC * 512);
        na1 = *(const v16h*)(wn + (size_t)1 * KC * 512);
        na2 = *(const v16h*)(wn + (size_t)2 * KC * 512);
        na3 = *(const v16h*)(wn + (size_t)3 * KC * 512);
      }
      SCHED_FENCE();  // keep prefetch loads above the WMMAs of chunk c
      acc[0] = __builtin_amdgcn_wmma_f32_16x16x32_f16(false, a0, false, bf, (short)0, acc[0], false, false);
      acc[1] = __builtin_amdgcn_wmma_f32_16x16x32_f16(false, a1, false, bf, (short)0, acc[1], false, false);
      acc[2] = __builtin_amdgcn_wmma_f32_16x16x32_f16(false, a2, false, bf, (short)0, acc[2], false, false);
      acc[3] = __builtin_amdgcn_wmma_f32_16x16x32_f16(false, a3, false, bf, (short)0, acc[3], false, false);
      if (c + 1 < KC) { bf = nbf; a0 = na0; a1 = na1; a2 = na2; a3 = na3; }
    }

    // C layout: lane<16 -> (co = tile*16 + r, pixel p); lane>=16 -> (co = +8+r)
    for (int m = 0; m < 4; ++m) {
      const int cobase = (cg * 4 + m) * 16 + ((lane < 16) ? 0 : 8);
      const v8f bv = *(const v8f*)(bias + (size_t)k * Co + cobase);
      for (int r = 0; r < 8; ++r)
        out[m][r] += wk * fmaxf(acc[m][r] + bv[r], 0.f);
    }
  }

  if (!IDX0 && no_act) {  // identity fallback (Ci==Co, stride==1 for these blocks)
    for (int m = 0; m < 4; ++m) {
      const int cobase = (cg * 4 + m) * 16 + ((lane < 16) ? 0 : 8);
      const v8h hv = *(const v8h*)(hin + (((size_t)b * Hin + y) * Win + x) * Ci + cobase);
      for (int r = 0; r < 8; ++r) out[m][r] = (float)hv[r];
    }
  }

  for (int m = 0; m < 4; ++m) {
    const int cobase = (cg * 4 + m) * 16 + ((lane < 16) ? 0 : 8);
    v8h st;
    for (int r = 0; r < 8; ++r) st[r] = (_Float16)out[m][r];
    *(v8h*)(hout + (((size_t)b * Hout + y) * Wout + x) * Co + cobase) = st;
  }
}

// ---------------------------------------------------------------------------
// Head: relu -> 4x4 avgpool -> flatten (C-major) ; then linear; then latency
// ---------------------------------------------------------------------------
__global__ void pool_kernel(const _Float16* __restrict__ h, float* __restrict__ pooled) {
  int id = blockIdx.x * blockDim.x + threadIdx.x;  // [B, 2048] flat (c*4+py*2+px)
  if (id >= BATCH * 2048) return;
  int f = id & 2047; int b = id >> 11;
  int c = f >> 2; int py = (f >> 1) & 1; int px = f & 1;
  float s = 0.f;
  for (int dy = 0; dy < 4; ++dy)
    for (int dx = 0; dx < 4; ++dx) {
      int yy = py * 4 + dy, xx = px * 4 + dx;
      float v = (float)h[(((size_t)b * 8 + yy) * 8 + xx) * 512 + c];
      s += fmaxf(v, 0.f);
    }
  pooled[id] = s * (1.f / 16.f);
}

__global__ void linear_kernel(const float* __restrict__ pooled, const float* __restrict__ lw,
                              const float* __restrict__ lb, float* __restrict__ outp) {
  int id = blockIdx.x * blockDim.x + threadIdx.x;
  if (id >= BATCH * 100) return;
  int n = id % 100; int b = id / 100;
  float acc = lb[n];
  const float* pp = pooled + (size_t)b * 2048;
  const float* wp = lw + (size_t)n * 2048;
  for (int f = 0; f < 2048; ++f) acc += pp[f] * wp[f];
  outp[(size_t)b * 100 + n] = acc;
}

__global__ void lat_kernel(const int* __restrict__ policy, const float* __restrict__ lats,
                           float* __restrict__ latout) {
  int b = threadIdx.x;
  if (b >= BATCH) return;
  const int idx0f[6] = {1, 0, 1, 0, 1, 0};
  float lat = 0.f;
  for (int t = 0; t < 6; ++t) {
    int any = 0; float d = 0.f;
    for (int k = 0; k < 4; ++k) {
      int a = policy[(b * 6 + t) * 4 + k];
      any |= a; d += (float)a * lats[t * 4 + k];
    }
    if (!any) d = idx0f[t] ? lats[t * 4 + 2] : 0.f;
    lat += d;
  }
  latout[b] = lat;
}

// ---------------------------------------------------------------------------
extern "C" void kernel_launch(void* const* d_in, const int* in_sizes, int n_in,
                              void* d_out, int out_size, void* d_ws, size_t ws_size,
                              hipStream_t stream) {
  const float* x       = (const float*)d_in[0];
  const int*   policy  = (const int*)d_in[1];
  const float* conv1_w = (const float*)d_in[2];
  const float* bn1_s   = (const float*)d_in[3];
  const float* bn1_b   = (const float*)d_in[4];
  const float* wsrc[6] = {(const float*)d_in[5],  (const float*)d_in[7],  (const float*)d_in[9],
                          (const float*)d_in[11], (const float*)d_in[13], (const float*)d_in[15]};
  const float* bsrc[6] = {(const float*)d_in[6],  (const float*)d_in[8],  (const float*)d_in[10],
                          (const float*)d_in[12], (const float*)d_in[14], (const float*)d_in[16]};
  const float* lats  = (const float*)d_in[17];
  const float* lin_w = (const float*)d_in[18];
  const float* lin_b = (const float*)d_in[19];

  static const int CiA[6] = {32, 128, 128, 256, 256, 512};
  static const int CoA[6] = {128, 128, 256, 256, 512, 512};

  char* ws = (char*)d_ws;
  const size_t HBUF = (size_t)BATCH * 32 * 32 * 128 * sizeof(_Float16);  // 16.8 MB
  _Float16* bufA = (_Float16*)ws;
  _Float16* bufB = (_Float16*)(ws + HBUF);
  size_t off = 2 * HBUF;
  _Float16* wpk[6];
  for (int t = 0; t < 6; ++t) {
    wpk[t] = (_Float16*)(ws + off);
    off += (size_t)4 * CoA[t] * CiA[t] * 9 * sizeof(_Float16);
    off = (off + 255) & ~(size_t)255;
  }
  float* pooled = (float*)(ws + off);

  // conv1 -> bufA (NHWC f16)
  {
    int n = BATCH * 32 * 32 * 32;
    conv1_kernel<<<(n + 255) / 256, 256, 0, stream>>>(x, conv1_w, bn1_s, bn1_b, bufA);
  }
  // pack all block weights into WMMA fragment layout
  for (int t = 0; t < 6; ++t) {
    long n = 4L * CoA[t] * CiA[t] * 9;
    pack_w_kernel<<<(int)((n + 255) / 256), 256, 0, stream>>>(wsrc[t], wpk[t], CoA[t], CiA[t]);
  }

  // fused WMMA blocks, ping-pong bufA/bufB
  auto grid_for = [](int Co, int Npix) { return ((Co / 64) * (Npix / 16) + 7) / 8; };

  // t=0: 32->128, 32x32 -> 32x32, s1, idx0
  block_gemm_kernel<32, 128, 32, 32, 32, 32, 1, 1>
      <<<grid_for(128, BATCH * 1024), 256, 0, stream>>>(bufA, wpk[0], bsrc[0], policy, bufB, 0);
  // t=1: 128->128, 32x32, s1
  block_gemm_kernel<128, 128, 32, 32, 32, 32, 1, 0>
      <<<grid_for(128, BATCH * 1024), 256, 0, stream>>>(bufB, wpk[1], bsrc[1], policy, bufA, 1);
  // t=2: 128->256, 32x32 -> 16x16, s2, idx0
  block_gemm_kernel<128, 256, 32, 32, 16, 16, 2, 1>
      <<<grid_for(256, BATCH * 256), 256, 0, stream>>>(bufA, wpk[2], bsrc[2], policy, bufB, 2);
  // t=3: 256->256, 16x16, s1
  block_gemm_kernel<256, 256, 16, 16, 16, 16, 1, 0>
      <<<grid_for(256, BATCH * 256), 256, 0, stream>>>(bufB, wpk[3], bsrc[3], policy, bufA, 3);
  // t=4: 256->512, 16x16 -> 8x8, s2, idx0
  block_gemm_kernel<256, 512, 16, 16, 8, 8, 2, 1>
      <<<grid_for(512, BATCH * 64), 256, 0, stream>>>(bufA, wpk[4], bsrc[4], policy, bufB, 4);
  // t=5: 512->512, 8x8, s1
  block_gemm_kernel<512, 512, 8, 8, 8, 8, 1, 0>
      <<<grid_for(512, BATCH * 64), 256, 0, stream>>>(bufB, wpk[5], bsrc[5], policy, bufA, 5);

  // head
  pool_kernel<<<(BATCH * 2048 + 255) / 256, 256, 0, stream>>>(bufA, pooled);
  linear_kernel<<<(BATCH * 100 + 255) / 256, 256, 0, stream>>>(pooled, lin_w, lin_b, (float*)d_out);
  lat_kernel<<<1, 64, 0, stream>>>(policy, lats, ((float*)d_out) + BATCH * 100);
}
